// HybridSelfAttention_65481071408743
// MI455X (gfx1250) — matrix operand order repaired and checked by reading, not compiled, not run on hardware
//
#include <hip/hip_runtime.h>
#include <hip/hip_bf16.h>

// ---------------------------------------------------------------------------
// HybridSelfAttention for MI455X (gfx1250), wave32 + WMMA bf16 path.
//   q,k,v = x@W^T + b      (bf16 WMMA, fp32 accum)
//   P     = softmax(q k^T / sqrt(D)) * rownorm(rot)^2
//   out   = P @ v          (bf16 WMMA, fp32 accum -> fp32 out)
// All GEMMs: 128x128x32 block tiles, 8 waves (4x2), double-buffered LDS with
// a 1-deep software pipeline; bf16->bf16 tile copies use the gfx1250 async
// global->LDS path (ASYNCcnt) when the toolchain exposes the builtin.
// ---------------------------------------------------------------------------

typedef __attribute__((ext_vector_type(16))) __bf16 v16bf;
typedef __attribute__((ext_vector_type(8)))  float  v8f;
typedef __attribute__((ext_vector_type(4)))  int    v4i;

constexpr int Bsz = 8;
constexpr int Ssz = 2048;
constexpr int Dsz = 1024;

constexpr int BM  = 128;
constexpr int BN  = 128;
constexpr int BK  = 32;
constexpr int LDT = BK + 8;       // padded LDS pitch (80B rows -> conflict-free)
constexpr int TILE = BM * LDT;    // elems per tile buffer
constexpr float SCALE = 0.03125f; // 1/sqrt(1024)

// ---- gfx1250 async global->LDS copy (guarded; sync fallback) ---------------
#if defined(__has_builtin)
#if __has_builtin(__builtin_amdgcn_global_load_async_to_lds_b128) && \
    __has_builtin(__builtin_amdgcn_s_wait_asynccnt)
#define HAVE_ASYNC_LDS 1
#endif
#endif
#ifndef HAVE_ASYNC_LDS
#define HAVE_ASYNC_LDS 0
#endif

__device__ inline void cp_async16(__bf16* l, const __bf16* g) {
#if HAVE_ASYNC_LDS
  __builtin_amdgcn_global_load_async_to_lds_b128(
      (__attribute__((address_space(1))) v4i*)g,
      (__attribute__((address_space(3))) v4i*)l, 0, 0);
#else
  *(uint4*)l = *(const uint4*)g;
#endif
}
__device__ inline void async_wait0() {
#if HAVE_ASYNC_LDS
  __builtin_amdgcn_s_wait_asynccnt(0);
#endif
}

__device__ inline v8f wmma_bf16(v16bf a, v16bf b, v8f c) {
  return __builtin_amdgcn_wmma_f32_16x16x32_bf16(
      false, a, false, b, (short)0, c, false, false);
}

// ---- per-lane WMMA fragment loads from padded LDS tiles --------------------
__device__ inline v16bf load_afrag(const __bf16* tile, int row0, int lane) {
  int m  = lane & 15;
  int kb = (lane >> 4) << 3;
  union { v16bf v; uint4 q[2]; } u;
  const __bf16* p = tile + (row0 + m) * LDT;
  u.q[0] = *(const uint4*)(p + kb);
  u.q[1] = *(const uint4*)(p + 16 + kb);
  return u.v;
}
__device__ inline v16bf load_bfrag(const __bf16* tile, int col0, int lane) {
  int n  = lane & 15;
  int ks = (lane >> 4) << 3;
  union { v16bf v; uint4 q[2]; } u;
  const __bf16* p = tile + (col0 + n) * LDT + ks;
  u.q[0] = *(const uint4*)(p);
  u.q[1] = *(const uint4*)(p + 16);
  return u.v;
}

__device__ inline void mma_step(const __bf16* As, const __bf16* Bs, int lane,
                                int aBase, int bBase, v8f acc[2][4]) {
  v16bf a0 = load_afrag(As, aBase,      lane);
  v16bf a1 = load_afrag(As, aBase + 16, lane);
#pragma unroll
  for (int j = 0; j < 4; ++j) {
    v16bf bf = load_bfrag(Bs, bBase + j * 16, lane);
    acc[0][j] = wmma_bf16(a0, bf, acc[0][j]);
    acc[1][j] = wmma_bf16(a1, bf, acc[1][j]);
  }
}

// ---- register-staged stores into LDS ---------------------------------------
__device__ inline void store16_f32(__bf16* l, float4 f0, float4 f1,
                                   float4 f2, float4 f3) {
  union { __bf16 e[16]; uint4 q[2]; } u;
  u.e[0]=(__bf16)f0.x; u.e[1]=(__bf16)f0.y; u.e[2]=(__bf16)f0.z; u.e[3]=(__bf16)f0.w;
  u.e[4]=(__bf16)f1.x; u.e[5]=(__bf16)f1.y; u.e[6]=(__bf16)f1.z; u.e[7]=(__bf16)f1.w;
  u.e[8]=(__bf16)f2.x; u.e[9]=(__bf16)f2.y; u.e[10]=(__bf16)f2.z; u.e[11]=(__bf16)f2.w;
  u.e[12]=(__bf16)f3.x; u.e[13]=(__bf16)f3.y; u.e[14]=(__bf16)f3.z; u.e[15]=(__bf16)f3.w;
  *(uint4*)l = u.q[0];
  *(uint4*)(l + 8) = u.q[1];
}
__device__ inline void store16_tr(__bf16* Bs, uint4 qa, uint4 qb, int tn, int tk) {
  union { __bf16 e[16]; uint4 q[2]; } u;
  u.q[0] = qa; u.q[1] = qb;
#pragma unroll
  for (int j = 0; j < 16; ++j) Bs[(tn + j) * LDT + tk] = u.e[j];
}

// ---------------------------------------------------------------------------
// Kernel 1: quantum mask = rownormalize(rot)^2, fp32, one block per row.
// ---------------------------------------------------------------------------
__global__ __launch_bounds__(256) void mask_kernel(const float* __restrict__ rot,
                                                   float* __restrict__ mask) {
  __shared__ float red[256];
  int r = blockIdx.x;
  int tid = threadIdx.x;
  const float* row = rot + (size_t)r * Ssz;
  float ss = 0.f;
  for (int c = tid; c < Ssz; c += 256) { float v = row[c]; ss += v * v; }
  red[tid] = ss;
  __syncthreads();
  for (int s = 128; s > 0; s >>= 1) {
    if (tid < s) red[tid] += red[tid + s];
    __syncthreads();
  }
  float d = fmaxf(sqrtf(red[0]), 1e-12f);
  float inv2 = 1.0f / (d * d);
  float* out = mask + (size_t)r * Ssz;
  for (int c = tid; c < Ssz; c += 256) { float v = row[c]; out[c] = v * v * inv2; }
}

// ---------------------------------------------------------------------------
// Kernel 2: QKV projection.  C[m,n] = sum_k x[m,k]*W[n,k] + b[n]  (bf16 out)
// fp32 sources -> register pipeline (needs cvt) -> LDS double buffer.
// grid = (M/128, D/128, 3)
// ---------------------------------------------------------------------------
__global__ __launch_bounds__(256) void qkv_kernel(
    const float* __restrict__ x,
    const float* __restrict__ Wq, const float* __restrict__ bq,
    const float* __restrict__ Wk, const float* __restrict__ bk,
    const float* __restrict__ Wv, const float* __restrict__ bv,
    __bf16* __restrict__ Qb, __bf16* __restrict__ Kb, __bf16* __restrict__ Vb) {
  __shared__ __bf16 As[2][TILE];
  __shared__ __bf16 Bs[2][TILE];

  const float* W    = (blockIdx.z == 0) ? Wq : (blockIdx.z == 1) ? Wk : Wv;
  const float* bias = (blockIdx.z == 0) ? bq : (blockIdx.z == 1) ? bk : bv;
  __bf16* out       = (blockIdx.z == 0) ? Qb : (blockIdx.z == 1) ? Kb : Vb;

  int tid  = threadIdx.x;
  int lane = tid & 31;
  int wave = tid >> 5;
  int aBase = (wave & 3) * 32;
  int bBase = (wave >> 2) * 64;
  int m0 = blockIdx.x * BM;
  int n0 = blockIdx.y * BN;
  int sRow = tid >> 1;
  int sCol = (tid & 1) * 16;

  const float* ag = x + (size_t)(m0 + sRow) * Dsz + sCol;
  const float* bg = W + (size_t)(n0 + sRow) * Dsz + sCol;
  __bf16* aL0 = &As[0][sRow * LDT + sCol];
  __bf16* aL1 = &As[1][sRow * LDT + sCol];
  __bf16* bL0 = &Bs[0][sRow * LDT + sCol];
  __bf16* bL1 = &Bs[1][sRow * LDT + sCol];

  v8f acc[2][4];
#pragma unroll
  for (int i = 0; i < 2; ++i)
#pragma unroll
    for (int j = 0; j < 4; ++j) acc[i][j] = (v8f){0, 0, 0, 0, 0, 0, 0, 0};

  // prologue: load tile 0 into registers
  float4 a0, a1, a2, a3, b0, b1, b2, b3;
  {
    const float4* p = (const float4*)ag;
    const float4* q = (const float4*)bg;
    a0 = p[0]; a1 = p[1]; a2 = p[2]; a3 = p[3];
    b0 = q[0]; b1 = q[1]; b2 = q[2]; b3 = q[3];
  }

  int buf = 0;
  for (int k0 = 0; k0 < Dsz; k0 += BK) {
    store16_f32(buf ? aL1 : aL0, a0, a1, a2, a3);
    store16_f32(buf ? bL1 : bL0, b0, b1, b2, b3);
    __syncthreads();
    if (k0 + BK < Dsz) {
      const float4* p = (const float4*)(ag + k0 + BK);
      const float4* q = (const float4*)(bg + k0 + BK);
      __builtin_prefetch(ag + k0 + 2 * BK, 0, 1);
      __builtin_prefetch(bg + k0 + 2 * BK, 0, 1);
      a0 = p[0]; a1 = p[1]; a2 = p[2]; a3 = p[3];
      b0 = q[0]; b1 = q[1]; b2 = q[2]; b3 = q[3];
    }
    mma_step(&As[buf][0], &Bs[buf][0], lane, aBase, bBase, acc);
    buf ^= 1;
  }

  int nl = lane & 15;
  int mh = (lane >> 4) * 8;
#pragma unroll
  for (int i = 0; i < 2; ++i)
#pragma unroll
    for (int j = 0; j < 4; ++j) {
      int col = n0 + bBase + j * 16 + nl;
      float bv_ = bias[col];
#pragma unroll
      for (int r = 0; r < 8; ++r) {
        int row = m0 + aBase + i * 16 + mh + r;
        out[(size_t)row * Dsz + col] = (__bf16)(acc[i][j][r] + bv_);
      }
    }
}

// ---------------------------------------------------------------------------
// Kernel 3: scores[b] = (Q[b] @ K[b]^T) * scale  (bf16), grid=(S/128,S/128,B)
// Both operands bf16 -> async global->LDS copies, double buffered.
// ---------------------------------------------------------------------------
__global__ __launch_bounds__(256) void scores_kernel(
    const __bf16* __restrict__ Qb, const __bf16* __restrict__ Kb,
    __bf16* __restrict__ P) {
  __shared__ __bf16 As[2][TILE];
  __shared__ __bf16 Bs[2][TILE];

  int tid  = threadIdx.x;
  int lane = tid & 31;
  int wave = tid >> 5;
  int aBase = (wave & 3) * 32;
  int bBase = (wave >> 2) * 64;
  int m0 = blockIdx.x * BM;
  int n0 = blockIdx.y * BN;
  size_t boff = (size_t)blockIdx.z * Ssz * Dsz;
  int sRow = tid >> 1;
  int sCol = (tid & 1) * 16;

  const __bf16* ag = Qb + boff + (size_t)(m0 + sRow) * Dsz + sCol;
  const __bf16* bg = Kb + boff + (size_t)(n0 + sRow) * Dsz + sCol;

  v8f acc[2][4];
#pragma unroll
  for (int i = 0; i < 2; ++i)
#pragma unroll
    for (int j = 0; j < 4; ++j) acc[i][j] = (v8f){0, 0, 0, 0, 0, 0, 0, 0};

  auto stage = [&](int k0, int b) {
    __bf16* al = &As[b][sRow * LDT + sCol];
    __bf16* bl = &Bs[b][sRow * LDT + sCol];
    cp_async16(al,     ag + k0);
    cp_async16(al + 8, ag + k0 + 8);
    cp_async16(bl,     bg + k0);
    cp_async16(bl + 8, bg + k0 + 8);
  };

  stage(0, 0);
  int buf = 0;
  for (int k0 = 0; k0 < Dsz; k0 += BK) {
    async_wait0();         // this thread's copies into buf are complete
    __syncthreads();       // all threads' copies visible; buf^1 reads retired
    if (k0 + BK < Dsz) stage(k0 + BK, buf ^ 1);   // overlap with mma below
    mma_step(&As[buf][0], &Bs[buf][0], lane, aBase, bBase, acc);
    buf ^= 1;
  }

  __bf16* out = P + (size_t)blockIdx.z * Ssz * Ssz;
  int nl = lane & 15;
  int mh = (lane >> 4) * 8;
#pragma unroll
  for (int i = 0; i < 2; ++i)
#pragma unroll
    for (int j = 0; j < 4; ++j) {
      int col = n0 + bBase + j * 16 + nl;
#pragma unroll
      for (int r = 0; r < 8; ++r) {
        int row = m0 + aBase + i * 16 + mh + r;
        out[(size_t)row * Ssz + col] = (__bf16)(acc[i][j][r] * SCALE);
      }
    }
}

// ---------------------------------------------------------------------------
// Kernel 4: in-place row softmax * mask over P (bf16), one block per row.
// ---------------------------------------------------------------------------
__global__ __launch_bounds__(256) void softmax_mask_kernel(
    __bf16* __restrict__ P, const float* __restrict__ mask) {
  __shared__ float red[256];
  int row = blockIdx.x;
  int q   = row & (Ssz - 1);
  int tid = threadIdx.x;
  __bf16* p = P + (size_t)row * Ssz;
  const float* mrow = mask + (size_t)q * Ssz;

  union { uint4 q4; __bf16 e[8]; } u;
  u.q4 = ((const uint4*)p)[tid];
  float vals[8];
#pragma unroll
  for (int i = 0; i < 8; ++i) vals[i] = (float)u.e[i];

  float mx = vals[0];
#pragma unroll
  for (int i = 1; i < 8; ++i) mx = fmaxf(mx, vals[i]);
  red[tid] = mx;
  __syncthreads();
  for (int s = 128; s > 0; s >>= 1) {
    if (tid < s) red[tid] = fmaxf(red[tid], red[tid + s]);
    __syncthreads();
  }
  mx = red[0];
  __syncthreads();

  float sm = 0.f;
#pragma unroll
  for (int i = 0; i < 8; ++i) { vals[i] = __expf(vals[i] - mx); sm += vals[i]; }
  red[tid] = sm;
  __syncthreads();
  for (int s = 128; s > 0; s >>= 1) {
    if (tid < s) red[tid] += red[tid + s];
    __syncthreads();
  }
  float inv = 1.0f / red[0];

#pragma unroll
  for (int i = 0; i < 8; ++i)
    u.e[i] = (__bf16)(vals[i] * inv * mrow[tid * 8 + i]);
  ((uint4*)p)[tid] = u.q4;
}

// ---------------------------------------------------------------------------
// Kernel 5: out[b] = P[b] @ V[b]  (fp32 out), grid=(S/128, D/128, B)
// A (P tiles): async copies.  B (V tiles): register pipeline + LDS transpose.
// ---------------------------------------------------------------------------
__global__ __launch_bounds__(256) void out_kernel(
    const __bf16* __restrict__ P, const __bf16* __restrict__ Vb,
    float* __restrict__ out) {
  __shared__ __bf16 As[2][TILE];
  __shared__ __bf16 Bs[2][TILE];

  int tid  = threadIdx.x;
  int lane = tid & 31;
  int wave = tid >> 5;
  int aBase = (wave & 3) * 32;
  int bBase = (wave >> 2) * 64;
  int m0 = blockIdx.x * BM;
  int n0 = blockIdx.y * BN;
  const __bf16* Pb = P  + (size_t)blockIdx.z * Ssz * Ssz;
  const __bf16* Vg = Vb + (size_t)blockIdx.z * Ssz * Dsz;

  int sRow = tid >> 1;
  int sCol = (tid & 1) * 16;
  int tk = tid >> 3;            // 0..31
  int tn = (tid & 7) * 16;      // 0..112

  const __bf16* ag = Pb + (size_t)(m0 + sRow) * Ssz + sCol;
  const __bf16* bg = Vg + (size_t)tk * Dsz + n0 + tn;   // + k0*Dsz

  v8f acc[2][4];
#pragma unroll
  for (int i = 0; i < 2; ++i)
#pragma unroll
    for (int j = 0; j < 4; ++j) acc[i][j] = (v8f){0, 0, 0, 0, 0, 0, 0, 0};

  auto stageA = [&](int k0, int b) {
    __bf16* al = &As[b][sRow * LDT + sCol];
    cp_async16(al,     ag + k0);
    cp_async16(al + 8, ag + k0 + 8);
  };

  // prologue
  stageA(0, 0);
  uint4 va, vb;
  {
    const uint4* g4 = (const uint4*)bg;
    va = g4[0]; vb = g4[1];
  }

  int buf = 0;
  for (int k0 = 0; k0 < Ssz; k0 += BK) {
    async_wait0();                      // A copies into buf complete
    store16_tr(&Bs[buf][0], va, vb, tn, tk);
    __syncthreads();
    if (k0 + BK < Ssz) {
      stageA(k0 + BK, buf ^ 1);
      const uint4* g4 = (const uint4*)(bg + (size_t)(k0 + BK) * Dsz);
      va = g4[0]; vb = g4[1];
    }
    mma_step(&As[buf][0], &Bs[buf][0], lane, aBase, bBase, acc);
    buf ^= 1;
  }

  int nl = lane & 15;
  int mh = (lane >> 4) * 8;
  size_t obase = (size_t)blockIdx.z * Ssz * Dsz;
#pragma unroll
  for (int i = 0; i < 2; ++i)
#pragma unroll
    for (int j = 0; j < 4; ++j) {
      int col = n0 + bBase + j * 16 + nl;
#pragma unroll
      for (int r = 0; r < 8; ++r) {
        int row = m0 + aBase + i * 16 + mh + r;
        out[obase + (size_t)row * Dsz + col] = acc[i][j][r];
      }
    }
}

// ---------------------------------------------------------------------------
extern "C" void kernel_launch(void* const* d_in, const int* in_sizes, int n_in,
                              void* d_out, int out_size, void* d_ws, size_t ws_size,
                              hipStream_t stream) {
  const float* x   = (const float*)d_in[0];
  const float* Wq  = (const float*)d_in[1];
  const float* bq  = (const float*)d_in[2];
  const float* Wk  = (const float*)d_in[3];
  const float* bk  = (const float*)d_in[4];
  const float* Wv  = (const float*)d_in[5];
  const float* bv  = (const float*)d_in[6];
  const float* rot = (const float*)d_in[7];
  float* out = (float*)d_out;

  char* ws = (char*)d_ws;
  const size_t szQKV = (size_t)Bsz * Ssz * Dsz * sizeof(__bf16); // 32 MB each
  __bf16* Qb   = (__bf16*)(ws);
  __bf16* Kb   = (__bf16*)(ws + szQKV);
  __bf16* Vb   = (__bf16*)(ws + 2 * szQKV);
  float*  mask = (float*)(ws + 3 * szQKV);                       // 16 MB
  __bf16* P    = (__bf16*)(ws + 3 * szQKV + (size_t)Ssz * Ssz * sizeof(float)); // 64 MB

  mask_kernel<<<Ssz, 256, 0, stream>>>(rot, mask);

  dim3 gq(Bsz * Ssz / BM, Dsz / BN, 3);
  qkv_kernel<<<gq, 256, 0, stream>>>(x, Wq, bq, Wk, bk, Wv, bv, Qb, Kb, Vb);

  dim3 gs(Ssz / BM, Ssz / BN, Bsz);
  scores_kernel<<<gs, 256, 0, stream>>>(Qb, Kb, P);

  softmax_mask_kernel<<<Bsz * Ssz, 256, 0, stream>>>(P, mask);

  dim3 go(Ssz / BM, Dsz / BN, Bsz);
  out_kernel<<<go, 256, 0, stream>>>(P, Vb, out);
}
